// KnowledgeGraphEmbedder_38474317038207
// MI455X (gfx1250) — matrix operand order repaired and checked by reading, not compiled, not run on hardware
//
#include <hip/hip_runtime.h>
#include <hip/hip_bf16.h>

typedef __attribute__((ext_vector_type(16))) __bf16 v16bf;
typedef __attribute__((ext_vector_type(8)))  float  v8f;
typedef __attribute__((ext_vector_type(4)))  unsigned int u32x4;
typedef __attribute__((ext_vector_type(8)))  int i32x8;
typedef __attribute__((ext_vector_type(4)))  int i32x4;

union ABFrag { uint4 q[2]; v16bf v; };

// LDS row stride: 32 bf16 (64B data) + 16B pad -> 80B. Conflict-free b128 frags.
#define RS 40   /* ushorts per LDS row */

__device__ __forceinline__ unsigned short f32_to_bf16(float f) {
  union { float f; unsigned int u; } x; x.f = f;
  unsigned int u = x.u + 0x7FFFu + ((x.u >> 16) & 1u);   // round-to-nearest-even
  return (unsigned short)(u >> 16);
}

__device__ __forceinline__ void load4_bf16(const float* p, unsigned short* d) {
  float4 f = *(const float4*)p;
  d[0] = f32_to_bf16(f.x); d[1] = f32_to_bf16(f.y);
  d[2] = f32_to_bf16(f.z); d[3] = f32_to_bf16(f.w);
}

__device__ __forceinline__ unsigned lds_off(const void* p) {
  return (unsigned)(size_t)p;   // low 32 bits of flat LDS address = LDS byte offset
}

// ---------------------------------------------------------------------------
// Tensor Data Mover: 2D bf16 tile load Global -> LDS.
//   tile_d0 (contiguous k) x tile_d1 rows; row stride = stride0 elements.
//   pad_enable: +4 DWORDs LDS pad per 16 DWORDs (64B) stored  -> 80B LDS rows.
// D# bit packing per CDNA5 ISA.
// ---------------------------------------------------------------------------
__device__ __forceinline__ void tdm_load_tile(unsigned lds_byte, const void* gptr,
                                              unsigned tensor_d0, unsigned tensor_d1,
                                              unsigned tile_d0, unsigned tile_d1,
                                              unsigned long long stride0) {
  unsigned long long ga = (unsigned long long)gptr;
  u32x4 g0;
  g0[0] = 1u;                                               // count=1, user mode
  g0[1] = lds_byte;                                         // lds_addr
  g0[2] = (unsigned)ga;                                     // global_addr[31:0]
  g0[3] = (unsigned)((ga >> 32) & 0x01FFFFFFu) | (2u << 30);// addr[56:32] | type=2
  i32x8 g1;
  g1[0] = (int)((1u << 16) | (1u << 20) | (3u << 22) | (3u << 25));
          // data_size=2B | pad_enable | pad_interval=16DW | pad_amount=4DW
  g1[1] = (int)(tensor_d0 << 16);                           // tensor_dim0[15:0]
  g1[2] = (int)((tensor_d0 >> 16) | (tensor_d1 << 16));     // d0[31:16] | d1[15:0]
  g1[3] = (int)((tensor_d1 >> 16) | (tile_d0 << 16));       // d1[31:16] | tile_dim0
  g1[4] = (int)tile_d1;                                     // tile_dim1 (tile_dim2=0)
  g1[5] = (int)(unsigned)(stride0 & 0xFFFFFFFFu);           // dim0_stride[31:0]
  g1[6] = (int)(unsigned)((stride0 >> 32) & 0xFFFFu);       // dim0_stride[47:32]
  g1[7] = 0;
  i32x4 z4 = {0, 0, 0, 0};
#if __clang_major__ >= 23
  i32x8 z8 = {0, 0, 0, 0, 0, 0, 0, 0};
  __builtin_amdgcn_tensor_load_to_lds(g0, g1, z4, z4, z8, 0);
#else
  __builtin_amdgcn_tensor_load_to_lds(g0, g1, z4, z4, 0);
#endif
}

// ---------------------------------------------------------------------------
// Weight pre-pass: transpose f32 [K][N] -> bf16 [N][Kp], zero-padded K -> Kp.
// ---------------------------------------------------------------------------
__global__ __launch_bounds__(256)
void transpose_bf16_kernel(const float* __restrict__ In, unsigned short* __restrict__ Out,
                           int K, int N, int Kp, size_t inBatch, size_t outBatch) {
  __shared__ float tile[32][33];
  const float* in = In + (size_t)blockIdx.z * inBatch;
  unsigned short* out = Out + (size_t)blockIdx.z * outBatch;
  const int k0 = blockIdx.x * 32;
  const int n0 = blockIdx.y * 32;
  const int tx = threadIdx.x & 31, ty = threadIdx.x >> 5;
#pragma unroll
  for (int i = 0; i < 4; ++i) {
    const int k = k0 + ty + 8 * i;
    float v = 0.f;
    if (k < K && (n0 + tx) < N) v = in[(size_t)k * N + n0 + tx];
    tile[ty + 8 * i][tx] = v;
  }
  __syncthreads();
#pragma unroll
  for (int i = 0; i < 4; ++i) {
    const int n = n0 + ty + 8 * i;
    const int kk = k0 + tx;
    if (n < N && kk < Kp) out[(size_t)n * Kp + kk] = f32_to_bf16(tile[tx][ty + 8 * i]);
  }
}

// ---------------------------------------------------------------------------
// Tiled GEMM:  C[M,N] = act(A[M,K] @ W[K,N] + bias),  M = 16384.
// BM=128, BN=64, BK=32; 256 threads = 8 wave32s; wave w owns rows w*16..+16.
// B panel always TDM-loaded from pre-transposed bf16 Wt[N][Kp].
// A panel: TDM from bf16 activations (A_TDM) or manual f32->bf16 staging.
// ---------------------------------------------------------------------------
template <bool A_TDM, bool RELU, bool OUT_BF16>
__global__ __launch_bounds__(256, 2)
void gemm_bf16_kernel(const void* __restrict__ A, int lda, int Kreal,
                      const unsigned short* __restrict__ Wt, int Kp, int N,
                      const float* __restrict__ bias,
                      void* __restrict__ Cout, int ldc) {
  __shared__ __align__(16) unsigned short As[2][128 * RS];
  __shared__ __align__(16) unsigned short Bs[2][64 * RS];

  const int tid   = threadIdx.x;
  const int mBase = blockIdx.x * 128;
  const int nBase = blockIdx.y * 64;

  const float* Af = (const float*)A;
  const unsigned short* Ah = (const unsigned short*)A;

  // manual A staging coords (layer-1 f32 path)
  const int acg = tid & 7, arow0 = tid >> 3;
  unsigned short aReg[16];

  auto a_load = [&](int kBase) {
#pragma unroll
    for (int r = 0; r < 4; ++r) {
      const int gk0 = kBase + acg * 4;
      if (gk0 < Kreal) {
        load4_bf16(&Af[(size_t)(mBase + arow0 + 32 * r) * lda + gk0], &aReg[r * 4]);
      } else {
        aReg[r*4+0] = 0; aReg[r*4+1] = 0; aReg[r*4+2] = 0; aReg[r*4+3] = 0;
      }
    }
  };
  auto a_commit = [&](int buf) {
#pragma unroll
    for (int r = 0; r < 4; ++r)
#pragma unroll
      for (int i = 0; i < 4; ++i)
        As[buf][(arow0 + 32 * r) * RS + acg * 4 + i] = aReg[r * 4 + i];
  };

  auto issue_tiles = [&](int kBase, int buf) {   // wave0/lane0 only
    tdm_load_tile(lds_off(&Bs[buf][0]),
                  Wt + (size_t)nBase * Kp + kBase, Kp, N, 32, 64, Kp);
    if (A_TDM)
      tdm_load_tile(lds_off(&As[buf][0]),
                    Ah + (size_t)mBase * lda + kBase, lda, 16384, 32, 128, lda);
  };

  const int w = tid >> 5, lane = tid & 31, m = lane & 15, hl = lane >> 4;

  const v8f vzero = {0.f,0.f,0.f,0.f,0.f,0.f,0.f,0.f};
  v8f acc[4];
#pragma unroll
  for (int t = 0; t < 4; ++t) acc[t] = vzero;

  auto compute = [&](int buf) {
    ABFrag fa;
    const unsigned short* ap = &As[buf][(w * 16 + m) * RS];
    fa.q[0] = *(const uint4*)(ap + hl * 8);
    fa.q[1] = *(const uint4*)(ap + 16 + hl * 8);
#pragma unroll
    for (int t = 0; t < 4; ++t) {
      ABFrag fb;
      const unsigned short* bp = &Bs[buf][(t * 16 + m) * RS];
      fb.q[0] = *(const uint4*)(bp + hl * 8);
      fb.q[1] = *(const uint4*)(bp + 16 + hl * 8);
      acc[t] = __builtin_amdgcn_wmma_f32_16x16x32_bf16(
          false, fa.v, false, fb.v, (short)0, acc[t], false, false);
    }
  };

  const int nk = Kp >> 5;
  if (tid == 0) issue_tiles(0, 0);
  if (!A_TDM) { a_load(0); a_commit(0); }
  if (tid == 0) __builtin_amdgcn_s_wait_tensorcnt(0);
  __syncthreads();

  for (int kt = 0; kt < nk; ++kt) {
    const int  cur = kt & 1, nxt = cur ^ 1;
    const bool hasNext = (kt + 1) < nk;
    if (hasNext && tid == 0) issue_tiles((kt + 1) * 32, nxt);  // DMA overlaps WMMA
    if (hasNext && !A_TDM) a_load((kt + 1) * 32);
    compute(cur);
    if (hasNext && !A_TDM) a_commit(nxt);
    if (hasNext && tid == 0) __builtin_amdgcn_s_wait_tensorcnt(0);
    __syncthreads();
  }

#pragma unroll
  for (int t = 0; t < 4; ++t) {
    const int col = nBase + t * 16 + m;
    const float bv = bias ? bias[col] : 0.f;
#pragma unroll
    for (int v = 0; v < 8; ++v) {
      float x = acc[t][v] + bv;
      if (RELU) x = x > 0.f ? x : 0.f;
      const int row = mBase + w * 16 + v + 8 * hl;
      if (OUT_BF16)
        ((unsigned short*)Cout)[(size_t)row * ldc + col] = f32_to_bf16(x);
      else
        ((float*)Cout)[(size_t)row * ldc + col] = x;
    }
  }
}

// ---------------------------------------------------------------------------
// LayerNorm(1024) + ReLU, in place on f32 rows. One 256-thread block per row.
// ---------------------------------------------------------------------------
__global__ __launch_bounds__(256)
void ln_relu_kernel(float* __restrict__ H, const float* __restrict__ g,
                    const float* __restrict__ b) {
  const int row = blockIdx.x;
  const int tid = threadIdx.x;
  float4 x = *(const float4*)&H[(size_t)row * 1024 + tid * 4];
  float s = x.x + x.y + x.z + x.w;
  float q = x.x * x.x + x.y * x.y + x.z * x.z + x.w * x.w;
#pragma unroll
  for (int off = 16; off > 0; off >>= 1) {
    s += __shfl_xor(s, off, 32);
    q += __shfl_xor(q, off, 32);
  }
  __shared__ float ss[8], sq[8];
  if ((tid & 31) == 0) { ss[tid >> 5] = s; sq[tid >> 5] = q; }
  __syncthreads();
  float S = 0.f, Q = 0.f;
#pragma unroll
  for (int i = 0; i < 8; ++i) { S += ss[i]; Q += sq[i]; }
  const float mu  = S * (1.f / 1024.f);
  const float var = Q * (1.f / 1024.f) - mu * mu;
  const float rs  = __frsqrt_rn(var + 1e-5f);
  const int c0 = tid * 4;
  float4 y;
  y.x = fmaxf((x.x - mu) * rs * g[c0 + 0] + b[c0 + 0], 0.f);
  y.y = fmaxf((x.y - mu) * rs * g[c0 + 1] + b[c0 + 1], 0.f);
  y.z = fmaxf((x.z - mu) * rs * g[c0 + 2] + b[c0 + 2], 0.f);
  y.w = fmaxf((x.w - mu) * rs * g[c0 + 3] + b[c0 + 3], 0.f);
  *(float4*)&H[(size_t)row * 1024 + tid * 4] = y;
}

// ---------------------------------------------------------------------------
// Label-routed experts: out[b,:] = h[b,:] @ e_w[label[b]] + e_b[label[b]].
// 64x64 tile, 4 expert B-panels TDM-loaded per K-step, per-row select.
// Wave w: rows (w&3)*16..+16, cols (w>>2)*32..+32.
// ---------------------------------------------------------------------------
__global__ __launch_bounds__(256, 2)
void expert_kernel(const float* __restrict__ H,          // [B,1024] post LN/ReLU
                   const unsigned short* __restrict__ EWt,// [4][1024(n)][1024(k)] bf16
                   const float* __restrict__ EB,          // [4,1024]
                   const int* __restrict__ labels,
                   float* __restrict__ Out) {             // [B,1024]
  __shared__ __align__(16) unsigned short As[2][64 * RS];
  __shared__ __align__(16) unsigned short Bs[2][4][64 * RS];

  const int tid   = threadIdx.x;
  const int mBase = blockIdx.x * 64;
  const int nBase = blockIdx.y * 64;

  const int acg = tid & 7, arow0 = tid >> 3;
  unsigned short aReg[8];

  auto a_load = [&](int kBase) {
#pragma unroll
    for (int r = 0; r < 2; ++r)
      load4_bf16(&H[(size_t)(mBase + arow0 + 32 * r) * 1024 + kBase + acg * 4],
                 &aReg[r * 4]);
  };
  auto a_commit = [&](int buf) {
#pragma unroll
    for (int r = 0; r < 2; ++r)
#pragma unroll
      for (int i = 0; i < 4; ++i)
        As[buf][(arow0 + 32 * r) * RS + acg * 4 + i] = aReg[r * 4 + i];
  };

  auto issue_tiles = [&](int kBase, int buf) {   // wave0/lane0 only
#pragma unroll
    for (int e = 0; e < 4; ++e)
      tdm_load_tile(lds_off(&Bs[buf][e][0]),
                    EWt + (size_t)e * 1024 * 1024 + (size_t)nBase * 1024 + kBase,
                    1024, 1024, 32, 64, 1024);
  };

  const int w = tid >> 5, lane = tid & 31, m = lane & 15, hl = lane >> 4;
  const int rT = w & 3, cH = w >> 2;

  const v8f vzero = {0.f,0.f,0.f,0.f,0.f,0.f,0.f,0.f};
  v8f acc[4][2];
#pragma unroll
  for (int e = 0; e < 4; ++e)
#pragma unroll
    for (int t = 0; t < 2; ++t) acc[e][t] = vzero;

  auto compute = [&](int buf) {
    ABFrag fa;
    const unsigned short* ap = &As[buf][(rT * 16 + m) * RS];
    fa.q[0] = *(const uint4*)(ap + hl * 8);
    fa.q[1] = *(const uint4*)(ap + 16 + hl * 8);
#pragma unroll
    for (int t = 0; t < 2; ++t) {
      const int n = cH * 32 + t * 16 + m;
#pragma unroll
      for (int e = 0; e < 4; ++e) {
        ABFrag fb;
        const unsigned short* bp = &Bs[buf][e][n * RS];
        fb.q[0] = *(const uint4*)(bp + hl * 8);
        fb.q[1] = *(const uint4*)(bp + 16 + hl * 8);
        acc[e][t] = __builtin_amdgcn_wmma_f32_16x16x32_bf16(
            false, fa.v, false, fb.v, (short)0, acc[e][t], false, false);
      }
    }
  };

  if (tid == 0) issue_tiles(0, 0);
  a_load(0); a_commit(0);
  if (tid == 0) __builtin_amdgcn_s_wait_tensorcnt(0);
  __syncthreads();

  for (int kt = 0; kt < 32; ++kt) {            // K = 1024
    const int  cur = kt & 1, nxt = cur ^ 1;
    const bool hasNext = kt < 31;
    if (hasNext && tid == 0) issue_tiles((kt + 1) * 32, nxt);
    if (hasNext) a_load((kt + 1) * 32);
    compute(cur);
    if (hasNext) a_commit(nxt);
    if (hasNext && tid == 0) __builtin_amdgcn_s_wait_tensorcnt(0);
    __syncthreads();
  }

#pragma unroll
  for (int t = 0; t < 2; ++t) {
    const int col = nBase + cH * 32 + t * 16 + m;
#pragma unroll
    for (int v = 0; v < 8; ++v) {
      const int row = mBase + rT * 16 + v + 8 * hl;
      const int lab = labels[row];
      float x = (lab == 0) ? acc[0][t][v]
              : (lab == 1) ? acc[1][t][v]
              : (lab == 2) ? acc[2][t][v]
                           : acc[3][t][v];
      x += EB[(size_t)lab * 1024 + col];
      Out[(size_t)row * 1024 + col] = x;
    }
  }
}

// ---------------------------------------------------------------------------
extern "C" void kernel_launch(void* const* d_in, const int* in_sizes, int n_in,
                              void* d_out, int out_size, void* d_ws, size_t ws_size,
                              hipStream_t stream) {
  const float* cnX   = (const float*)d_in[0];
  const float* cmX   = (const float*)d_in[1];
  const int*   labels= (const int*)  d_in[2];
  const float* cn_w1 = (const float*)d_in[3];
  const float* cn_b1 = (const float*)d_in[4];
  const float* cn_w2 = (const float*)d_in[5];
  const float* cn_b2 = (const float*)d_in[6];
  const float* cm_w1 = (const float*)d_in[7];
  const float* cm_b1 = (const float*)d_in[8];
  const float* cm_w2 = (const float*)d_in[9];
  const float* cm_b2 = (const float*)d_in[10];
  const float* f_w   = (const float*)d_in[11];
  const float* f_b   = (const float*)d_in[12];
  const float* ln_w  = (const float*)d_in[13];
  const float* ln_b  = (const float*)d_in[14];
  const float* e_w   = (const float*)d_in[15];
  const float* e_b   = (const float*)d_in[16];

  // ---- workspace layout ----------------------------------------------------
  // [0, 11 MiB)      transposed bf16 weights (padded K)
  // [11, 27 MiB)     fused bf16 [B,512]
  // [27, 59 MiB)     Y1cn / Y1cm bf16 [B,512] each
  // [27, 91 MiB)     hpre f32 [B,1024]   (reuses dead Y1 region)
  char* ws = (char*)d_ws;
  unsigned short* w1t_cn = (unsigned short*)ws;                       // 512*224
  unsigned short* w1t_cm = w1t_cn + 512 * 224;
  unsigned short* w2t_cn = w1t_cm + 512 * 224;                       // 256*512
  unsigned short* w2t_cm = w2t_cn + 256 * 512;
  unsigned short* f_wt   = w2t_cm + 256 * 512;                       // 1024*512
  unsigned short* e_wt   = f_wt   + 1024 * 512;                      // 4*1024*1024
  const size_t WEND = 11u * 1024 * 1024;                             // 11 MiB
  unsigned short* fused = (unsigned short*)(ws + WEND);
  unsigned short* Y1cn  = (unsigned short*)(ws + WEND + 16777216);
  unsigned short* Y1cm  = (unsigned short*)(ws + WEND + 2 * 16777216);
  float*          hpre  = (float*)(ws + WEND + 16777216);

  const dim3 blk(256);

  // ---- weight pre-pass: f32 [K][N] -> bf16 transposed [N][Kp] --------------
  transpose_bf16_kernel<<<dim3(7, 16, 1),  blk, 0, stream>>>(cn_w1, w1t_cn, 200, 512, 224, 0, 0);
  transpose_bf16_kernel<<<dim3(7, 16, 1),  blk, 0, stream>>>(cm_w1, w1t_cm, 200, 512, 224, 0, 0);
  transpose_bf16_kernel<<<dim3(16, 8, 1),  blk, 0, stream>>>(cn_w2, w2t_cn, 512, 256, 512, 0, 0);
  transpose_bf16_kernel<<<dim3(16, 8, 1),  blk, 0, stream>>>(cm_w2, w2t_cm, 512, 256, 512, 0, 0);
  transpose_bf16_kernel<<<dim3(16, 32, 1), blk, 0, stream>>>(f_w,   f_wt, 512, 1024, 512, 0, 0);
  transpose_bf16_kernel<<<dim3(32, 32, 4), blk, 0, stream>>>(e_w,   e_wt, 1024, 1024, 1024,
                                                             1024 * 1024, 1024 * 1024);

  // ---- encoder layer 1: relu(X @ W1 + b1) -> bf16 [B,512] ------------------
  gemm_bf16_kernel<false, true, true><<<dim3(128, 8), blk, 0, stream>>>(
      cnX, 200, 200, w1t_cn, 224, 512, cn_b1, (void*)Y1cn, 512);
  gemm_bf16_kernel<false, true, true><<<dim3(128, 8), blk, 0, stream>>>(
      cmX, 200, 200, w1t_cm, 224, 512, cm_b1, (void*)Y1cm, 512);

  // ---- encoder layer 2 -> concatenated fused [B,512] -----------------------
  gemm_bf16_kernel<true, false, true><<<dim3(128, 4), blk, 0, stream>>>(
      Y1cn, 512, 512, w2t_cn, 512, 256, cn_b2, (void*)fused, 512);
  gemm_bf16_kernel<true, false, true><<<dim3(128, 4), blk, 0, stream>>>(
      Y1cm, 512, 512, w2t_cm, 512, 256, cm_b2, (void*)(fused + 256), 512);

  // ---- fusion: fused @ f_w + f_b -> f32 hpre [B,1024] ----------------------
  gemm_bf16_kernel<true, false, false><<<dim3(128, 16), blk, 0, stream>>>(
      fused, 512, 512, f_wt, 512, 1024, f_b, (void*)hpre, 1024);

  // ---- LayerNorm + ReLU in place -------------------------------------------
  ln_relu_kernel<<<dim3(16384), blk, 0, stream>>>(hpre, ln_w, ln_b);

  // ---- routed experts -> d_out f32 [B,1024] --------------------------------
  expert_kernel<<<dim3(256, 16), blk, 0, stream>>>(
      hpre, e_wt, e_b, labels, (float*)d_out);
}
